// SetConvGridEncoder_25194278159164
// MI455X (gfx1250) — matrix operand order repaired and checked by reading, not compiled, not run on hardware
//
#include <hip/hip_runtime.h>
#include <hip/hip_bf16.h>

// Problem constants from the reference
#define BATCH 4
#define NPTS  4096          // off-grid points (K dimension of the GEMM)
#define DX    2
#define DZ    128
#define GRID  64
#define MPTS  (GRID * GRID) // 4096 grid points (M dimension)

#define BM    128           // rows of output tile per block (4 waves x 32 rows)
#define KC    32            // K-chunk (== WMMA K for f16)
#define ASTR  40            // LDS row stride (halves): 80B, 16B-aligned, bank-skewed
#define BSTR  40

typedef __attribute__((ext_vector_type(16))) _Float16 v16h;
typedef __attribute__((ext_vector_type(8)))  _Float16 v8h;
typedef __attribute__((ext_vector_type(8)))  float    v8f;

__device__ inline v16h load_frag(const _Float16* p0, const _Float16* p1) {
    v8h lo = *(const v8h*)p0;   // ds_load_b128
    v8h hi = *(const v8h*)p1;   // ds_load_b128
    v16h r;
#pragma unroll
    for (int i = 0; i < 8; ++i) { r[i] = lo[i]; r[i + 8] = hi[i]; }
    return r;
}

__global__ __launch_bounds__(128) void setconv_wmma_kernel(
    const float* __restrict__ xc_off,   // [B, N, 2]
    const float* __restrict__ zc_off,   // [B, N, 128]
    const float* __restrict__ zc_on,    // [B, 64, 64, 128]
    const float* __restrict__ ls_param, // [2]
    float* __restrict__ out)            // [B*M*2] xc_on copy | [B, M, 128] z_grid
{
    __shared__ __align__(16) _Float16 As[BM * ASTR];   // weights tile, [m][k]
    __shared__ __align__(16) _Float16 Bs[DZ * BSTR];   // zc_off tile transposed, [d][k]

    const int b        = blockIdx.z;
    const int blockRow = blockIdx.x;           // 128-row tile index
    const int t        = threadIdx.x;
    const int wave     = t >> 5;
    const int lane     = t & 31;

    // lengthscale: 1e-5 + softplus(param), per ARD dim (scalarized by compiler)
    const float inv_ls0 = 1.0f / (1e-5f + log1pf(expf(ls_param[0])));
    const float inv_ls1 = 1.0f / (1e-5f + log1pf(expf(ls_param[1])));
    const float inv_g   = 1.0f / (float)(GRID - 1);    // linspace(0,1,G) step

    const float* xoff_b = xc_off + (size_t)b * NPTS * DX;
    const float* zoff_b = zc_off + (size_t)b * NPTS * DZ;

    // Each wave owns a 32x128 output strip: 2 row-tiles x 8 col-tiles of 16x16
    v8f acc[2][8] = {};

    // A-fragment addressing (16-bit A 16x32 layout):
    //  lanes 0-15 hold K 0-7 (vgpr0-3) and 16-23 (vgpr4-7); lanes 16-31 hold K 8-15, 24-31
    const int a_row0 = wave * 32 + (lane & 15);        // row-tile 0
    const int a_kb   = (lane < 16) ? 0 : 8;
    // B-fragment addressing (16-bit B 32x16): same K order as A (lanes 0-15 K 0-7,16-23; lanes 16-31 K 8-15,24-31)
    const int b_n    = lane & 15;
    const int b_kb   = (lane < 16) ? 0 : 8;

    // gx takes only two values per 128-row block: grid rows 2*blockRow, 2*blockRow+1
    const float gx0 = (float)(2 * blockRow) * inv_g;
    const float gx1 = gx0 + inv_g;

    for (int k0 = 0; k0 < NPTS; k0 += KC) {
        // ---- stage A (factorized RBF): w = exp(-.5 d0^2) * exp(-.5 d1^2) ----
        // kk = lane; per thread only 2 gx-exps + 16 gy-exps per chunk
        const float2 xy = ((const float2*)(xoff_b + (size_t)k0 * DX))[lane];
        const float d00 = (gx0 - xy.x) * inv_ls0;
        const float d01 = (gx1 - xy.x) * inv_ls0;
        const float ex0 = __expf(-0.5f * d00 * d00);
        const float ex1 = __expf(-0.5f * d01 * d01);
#pragma unroll
        for (int i = 0; i < 16; ++i) {
            const int m_lo = i * 4 + wave;                     // 0..63 (gy index)
            const float gy = (float)m_lo * inv_g;
            const float d1 = (gy - xy.y) * inv_ls1;
            const float ey = __expf(-0.5f * d1 * d1);
            As[m_lo * ASTR + lane]        = (_Float16)(ex0 * ey);  // rows 0..63
            As[(m_lo + 64) * ASTR + lane] = (_Float16)(ex1 * ey);  // rows 64..127
        }
        // ---- stage B: Bt[d][kk] = (f16) zc_off[b][k0+kk][d]; d = t, packed b128 ----
        const float* src = zoff_b + (size_t)k0 * DZ;
#pragma unroll
        for (int g = 0; g < 4; ++g) {
            v8h pack;
#pragma unroll
            for (int i = 0; i < 8; ++i)
                pack[i] = (_Float16)src[(g * 8 + i) * DZ + t]; // coalesced over t
            *(v8h*)&Bs[t * BSTR + g * 8] = pack;               // ds_store_b128
        }
        // prefetch next chunk's B data toward this WGP while we compute
        if (k0 + KC < NPTS) {
            __builtin_prefetch(src + KC * DZ + t * 32, 0, 0);  // global_prefetch_b8
        }
        __syncthreads();

        // ---- MMA: 32x128 strip per wave; each B fragment feeds 2 WMMAs ----
        const v16h a0 = load_frag(&As[a_row0 * ASTR + a_kb],
                                  &As[a_row0 * ASTR + a_kb + 16]);
        const v16h a1 = load_frag(&As[(a_row0 + 16) * ASTR + a_kb],
                                  &As[(a_row0 + 16) * ASTR + a_kb + 16]);
#pragma unroll
        for (int j = 0; j < 8; ++j) {
            const _Float16* bp = &Bs[(j * 16 + b_n) * BSTR + b_kb];
            const v16h bf = load_frag(bp, bp + 16);
            acc[0][j] = __builtin_amdgcn_wmma_f32_16x16x32_f16(
                false, a0, false, bf, (short)0, acc[0][j], false, false);
            acc[1][j] = __builtin_amdgcn_wmma_f32_16x16x32_f16(
                false, a1, false, bf, (short)0, acc[1][j], false, false);
        }
        __syncthreads();
    }

    // ---- epilogue: z_grid = zc_on_grid + acc ----
    // D layout: vgpr v -> row v (lanes 0-15) / v+8 (lanes 16-31); col = lane%16
    const float* zon_b = zc_on + (size_t)b * MPTS * DZ;
    float* outz        = out + (size_t)BATCH * MPTS * DX + (size_t)b * MPTS * DZ;
    const int col0 = lane & 15;
#pragma unroll
    for (int r = 0; r < 2; ++r) {
        const int rbase = blockRow * BM + wave * 32 + r * 16 + ((lane >> 4) << 3);
#pragma unroll
        for (int j = 0; j < 8; ++j) {
#pragma unroll
            for (int v = 0; v < 8; ++v) {
                const size_t idx = (size_t)(rbase + v) * DZ + (j * 16 + col0);
                outz[idx] = zon_b[idx] + acc[r][j][v];
            }
        }
    }
}

extern "C" void kernel_launch(void* const* d_in, const int* in_sizes, int n_in,
                              void* d_out, int out_size, void* d_ws, size_t ws_size,
                              hipStream_t stream) {
    const float* xc_off = (const float*)d_in[0];   // [B,N,2]
    const float* xc_on  = (const float*)d_in[1];   // [B,64,64,2]
    const float* zc_off = (const float*)d_in[2];   // [B,N,128]
    const float* zc_on  = (const float*)d_in[3];   // [B,64,64,128]
    const float* ls_p   = (const float*)d_in[4];   // [2]
    float* out = (float*)d_out;

    // Output 0: xc_on_grid passthrough (B*G*G*DX floats)
    hipMemcpyAsync(out, xc_on, (size_t)BATCH * MPTS * DX * sizeof(float),
                   hipMemcpyDeviceToDevice, stream);

    // Output 1: fused RBF-weight GEMM + residual add
    dim3 grid(MPTS / BM, 1, BATCH);   // 32 x 1 x 4 = 128 workgroups
    dim3 block(128);                  // 4 waves (wave32), each owns 32x128 of the tile
    setconv_wmma_kernel<<<grid, block, 0, stream>>>(xc_off, zc_off, zc_on, ls_p, out);
}
